// HeteroSAGE_40561671143575
// MI455X (gfx1250) — compile-verified, matrix operand order checked by reading, not run on hardware
//
#include <hip/hip_runtime.h>

#define D_DIM 256
#define LN_EPS 1e-5f

typedef __bf16 bf16;
typedef __attribute__((ext_vector_type(16))) __bf16 v16bf;
typedef __attribute__((ext_vector_type(8)))  float   v8f;

// D = A(16x32 bf16) * B(32x16 bf16) + C(16x16 f32), wave32, gfx1250
#define WMMA_BF16(A, B, C) \
  __builtin_amdgcn_wmma_f32_16x16x32_bf16(false, (A), false, (B), (short)0, (C), false, false)

// ---------------------------------------------------------------------------
// Edge scatter: one wave (32 lanes) per edge. Gather src row (coalesced),
// atomic-add into per-destination accumulator (102MB < 192MB L2 -> atomics
// resolve in L2) plus a float degree counter (matches segment_sum of ones).
// ---------------------------------------------------------------------------
__global__ __launch_bounds__(256)
void scatter_add_kernel(const float* __restrict__ x_src, const int* __restrict__ ei,
                        int E, float* __restrict__ s, float* __restrict__ cnt)
{
  int t    = blockIdx.x * blockDim.x + threadIdx.x;
  int edge = t >> 5;
  int lane = t & 31;
  if (edge >= E) return;
  int src = ei[edge];       // ei row 0 = source indices
  int dst = ei[E + edge];   // ei row 1 = destination indices
  if (lane == 0) atomicAdd(&cnt[dst], 1.0f);
  const float* xr = x_src + (size_t)src * D_DIM;
  float*       sr = s     + (size_t)dst * D_DIM;
#pragma unroll
  for (int j = 0; j < 8; ++j) {
    int col = lane + 32 * j;                 // consecutive lanes -> consecutive addrs
    atomicAdd(&sr[col], xr[col]);
  }
}

// ---------------------------------------------------------------------------
// Prepack one weight matrix (optionally W+W2) into bf16 hi/lo arrays stored in
// WMMA B-fragment order, so the GEMM can load v16bf fragments directly:
//   B frag (32x16): lane n = lane&15, group g = lane>>4, element e -> K = 8g + (e<8?e:e+8)
//   storage index: ((kblk*256 + ncol)*32 + g*16 + e)
// Also pre-sums the bias (bl[l,1]+bl[l,2] for the user update).
// ---------------------------------------------------------------------------
__global__ __launch_bounds__(256)
void prepack_w_kernel(const float* __restrict__ W, const float* __restrict__ W2,
                      const float* __restrict__ bias, const float* __restrict__ bias2,
                      bf16* __restrict__ Bhi, bf16* __restrict__ Blo,
                      float* __restrict__ bias_out)
{
  int idx = blockIdx.x * 256 + threadIdx.x;   // 0..65535 over [ncol=256][k=256]
  float w = W[idx];
  if (W2) w += W2[idx];
  int ncol = idx >> 8;
  int k    = idx & 255;
  int kblk = k >> 5;
  int kr   = k & 31;
  // invert frag map: kr in [0,8):g0,e=kr  [8,16):g1,e=kr-8  [16,24):g0,e=kr-8  [24,32):g1,e=kr-16
  int g = (kr >> 3) & 1;
  int e = (kr & 7) + ((kr >> 4) << 3);
  int o = (kblk * 256 + ncol) * 32 + g * 16 + e;
  bf16 h = (bf16)w;
  Bhi[o] = h;
  Blo[o] = (bf16)(w - (float)h);
  if (bias_out && idx < D_DIM) {
    float b = bias[idx];
    if (bias2) b += bias2[idx];
    bias_out[idx] = b;
  }
}

// ---------------------------------------------------------------------------
// Cooperative A-tile staging: 256 threads load the block's 16x256 A tile
// (coalesced b128s), apply per-row mean scale, split to bf16 hi/lo ONCE, and
// store into LDS in A-fragment order:
//   A frag (16x32): lane m = lane&15, group g = lane>>4, elem e -> K = 8g + (e<8?e:e+8)
//   LDS index: (kblk*32 + g*16 + m)*16 + e      (lane = g*16+m)
// ---------------------------------------------------------------------------
static __device__ __forceinline__ void stage_a_lds(
    const float* __restrict__ A, const float* __restrict__ cnt,
    int m_base, int n_rows,
    bf16* __restrict__ ahi, bf16* __restrict__ alo, float* __restrict__ sscale)
{
  __syncthreads();                              // prior readers of LDS done
  if (threadIdx.x < 16) {
    int row = m_base + (int)threadIdx.x;
    if (row >= n_rows) row = n_rows - 1;
    sscale[threadIdx.x] = cnt ? (1.0f / fmaxf(cnt[row], 1.0f)) : 1.0f;
  }
  __syncthreads();
  const int m    = threadIdx.x & 15;
  const int col0 = (threadIdx.x >> 4) * 16;     // thread owns A[row][col0..col0+15]
  int row = m_base + m;
  if (row >= n_rows) row = n_rows - 1;          // clamp: keeps EXEC all-ones
  const float* ar = A + (size_t)row * D_DIM + col0;
  const float sc = sscale[m];
  float v[16];
#pragma unroll
  for (int j = 0; j < 16; ++j) v[j] = ar[j] * sc;
  const int kblk = col0 >> 5;
  const int half = (col0 >> 4) & 1;             // which 16-wide half of the 32-k block
#pragma unroll
  for (int g = 0; g < 2; ++g) {
    int base = (kblk * 32 + g * 16 + m) * 16 + half * 8;   // 8 contiguous bf16 (16B aligned)
#pragma unroll
    for (int q = 0; q < 8; ++q) {
      float x = v[g * 8 + q];
      bf16 h = (bf16)x;
      ahi[base + q] = h;
      alo[base + q] = (bf16)(x - (float)h);
    }
  }
  __syncthreads();
}

// ---------------------------------------------------------------------------
// One GEMM term: acc += bf16x3(A_tile) @ Wfrag^T.  Inner loop is pure
// ds_load_b128 (A frags) + global b128 (prepacked B frags) + 6 WMMA.
// ---------------------------------------------------------------------------
static __device__ __forceinline__ void gemm_compute(
    const bf16* __restrict__ ahi_l, const bf16* __restrict__ alo_l,
    const bf16* __restrict__ Bhi, const bf16* __restrict__ Blo,
    int n_base, int lane, v8f& acc0, v8f& acc1)
{
  const int n  = lane & 15;
  const int g  = lane >> 4;
  const int c0 = n_base + n;
  const int c1 = c0 + 16;
#pragma unroll
  for (int kblk = 0; kblk < 8; ++kblk) {
    v16bf ahi = *(const v16bf*)(ahi_l + (kblk * 32 + lane) * 16);
    v16bf alo = *(const v16bf*)(alo_l + (kblk * 32 + lane) * 16);
    size_t i0 = ((size_t)(kblk * 256 + c0) * 32) + (size_t)g * 16;
    size_t i1 = ((size_t)(kblk * 256 + c1) * 32) + (size_t)g * 16;
    v16bf b0hi = *(const v16bf*)(Bhi + i0);
    v16bf b0lo = *(const v16bf*)(Blo + i0);
    v16bf b1hi = *(const v16bf*)(Bhi + i1);
    v16bf b1lo = *(const v16bf*)(Blo + i1);
    acc0 = WMMA_BF16(ahi, b0hi, acc0);
    acc0 = WMMA_BF16(ahi, b0lo, acc0);
    acc0 = WMMA_BF16(alo, b0hi, acc0);
    acc1 = WMMA_BF16(ahi, b1hi, acc1);
    acc1 = WMMA_BF16(ahi, b1lo, acc1);
    acc1 = WMMA_BF16(alo, b1hi, acc1);
  }
}

// ---------------------------------------------------------------------------
// Fused:  out = relu(LN( sum_t meanOrDense(A_t) @ W_t^T + bias ))
// Block = 256 thr = 8 wave32; computes a 16-row x 256-col stripe. Wave w owns
// columns [32w,32w+32) as two WMMA tiles. LDS (16KB) holds the staged A tile
// during GEMM and is reused as the f32 tile for the LayerNorm epilogue.
// ---------------------------------------------------------------------------
__global__ __launch_bounds__(256)
void sage_gemm_ln_kernel(
    const float* __restrict__ A0, const float* __restrict__ cnt0,
    const bf16*  __restrict__ B0hi, const bf16* __restrict__ B0lo,
    const float* __restrict__ A1, const float* __restrict__ cnt1,
    const bf16*  __restrict__ B1hi, const bf16* __restrict__ B1lo,
    const float* __restrict__ A2,
    const bf16*  __restrict__ B2hi, const bf16* __restrict__ B2lo,
    const float* __restrict__ bias,
    const float* __restrict__ gamma, const float* __restrict__ beta,
    float* __restrict__ out, int n_rows)
{
  __shared__ float smem[16 * D_DIM];   // 16KB: bf16 hi/lo A staging, later f32 epilogue tile
  __shared__ float sscale[16];
  bf16* ahi_l = (bf16*)smem;
  bf16* alo_l = ahi_l + 16 * D_DIM;

  const int lane   = threadIdx.x & 31;
  const int wave   = threadIdx.x >> 5;
  const int m_base = blockIdx.x * 16;
  const int n_base = wave * 32;

  v8f acc0 = {}; v8f acc1 = {};
  stage_a_lds(A0, cnt0, m_base, n_rows, ahi_l, alo_l, sscale);
  gemm_compute(ahi_l, alo_l, B0hi, B0lo, n_base, lane, acc0, acc1);
  if (A1) {
    stage_a_lds(A1, cnt1, m_base, n_rows, ahi_l, alo_l, sscale);
    gemm_compute(ahi_l, alo_l, B1hi, B1lo, n_base, lane, acc0, acc1);
  }
  stage_a_lds(A2, nullptr, m_base, n_rows, ahi_l, alo_l, sscale);
  gemm_compute(ahi_l, alo_l, B2hi, B2lo, n_base, lane, acc0, acc1);
  __syncthreads();   // A staging consumed; smem becomes the f32 epilogue tile

  // C/D layout: VGPR r -> M = r + 8*(lane>>4), N = lane&15
  const int n = lane & 15;
  const int g = lane >> 4;
  float b0 = bias[n_base + n];
  float b1 = bias[n_base + 16 + n];
#pragma unroll
  for (int r = 0; r < 8; ++r) {
    int row = 8 * g + r;
    smem[row * D_DIM + n_base + n]      = acc0[r] + b0;
    smem[row * D_DIM + n_base + 16 + n] = acc1[r] + b1;
  }
  __syncthreads();

  // LayerNorm + ReLU: wave w handles rows w and w+8
#pragma unroll
  for (int rr = wave; rr < 16; rr += 8) {
    float v[8];
    float sum = 0.0f, sq = 0.0f;
#pragma unroll
    for (int j = 0; j < 8; ++j) {
      v[j] = smem[rr * D_DIM + lane + 32 * j];
      sum += v[j];
      sq  += v[j] * v[j];
    }
#pragma unroll
    for (int off = 16; off > 0; off >>= 1) {
      sum += __shfl_xor(sum, off, 32);
      sq  += __shfl_xor(sq,  off, 32);
    }
    float mu  = sum * (1.0f / D_DIM);
    float var = sq  * (1.0f / D_DIM) - mu * mu;
    float rs  = rsqrtf(var + LN_EPS);
    int grow = m_base + rr;
    if (grow < n_rows) {
#pragma unroll
      for (int j = 0; j < 8; ++j) {
        int col = lane + 32 * j;
        float y = (v[j] - mu) * rs * gamma[col] + beta[col];
        out[(size_t)grow * D_DIM + col] = fmaxf(y, 0.0f);
      }
    }
  }
}

// ---------------------------------------------------------------------------
// Driver
// ---------------------------------------------------------------------------
extern "C" void kernel_launch(void* const* d_in, const int* in_sizes, int n_in,
                              void* d_out, int out_size, void* d_ws, size_t ws_size,
                              hipStream_t stream) {
  (void)n_in; (void)out_size; (void)ws_size;
  const float* x_user = (const float*)d_in[0];
  const float* x_item = (const float*)d_in[1];
  const float* Wl     = (const float*)d_in[2];  // [L,3,D,D]
  const float* bl     = (const float*)d_in[3];  // [L,3,D]
  const float* Wr     = (const float*)d_in[4];  // [L,3,D,D]
  const float* ln_g   = (const float*)d_in[5];  // [L,2,D]
  const float* ln_b   = (const float*)d_in[6];  // [L,2,D]
  const int* ei_ui    = (const int*)d_in[7];    // [2,E] user->item
  const int* ei_iu    = (const int*)d_in[8];    // [2,E] item->user
  const int* ei_uu    = (const int*)d_in[9];    // [2,E] user->user

  const int D  = D_DIM;
  const int NU = in_sizes[0] / D;
  const int NI = in_sizes[1] / D;
  const int E  = in_sizes[7] / 2;
  const int Nmax = (NU > NI) ? NU : NI;
  const size_t NDmax = (size_t)Nmax * D;
  const size_t WMAT  = (size_t)D * D;   // 65536
  const int L = 2;

  // ---- workspace layout ----
  float* ws    = (float*)d_ws;
  float* s_a   = ws;                      // [Nmax, D] segment-sum accumulator A
  float* s_b   = s_a + NDmax;             // [Nmax, D] segment-sum accumulator B
  float* cnt_a = s_b + NDmax;             // [Nmax]
  float* cnt_b = cnt_a + Nmax;            // [Nmax]
  float* h1_u  = cnt_b + Nmax;            // [NU, D] inter-layer user feats
  float* h1_i  = h1_u + (size_t)NU * D;   // [NI, D]
  uintptr_t pp = (uintptr_t)(h1_i + (size_t)NI * D);
  pp = (pp + 31) & ~(uintptr_t)31;        // 32B align for v16bf loads
  bf16* pw = (bf16*)pp;                   // 10 prepacked matrices (hi+lo each 65536 bf16)
  const size_t PM = 2 * WMAT;             // bf16 elems per matrix slot
  float* pbias = (float*)(pw + 10 * PM);  // [2L, D] pre-summed biases

  float* out_u_final = (float*)d_out;
  float* out_i_final = (float*)d_out + (size_t)NU * D;

  // ---- prepack all weights/biases once (independent of graph data) ----
  for (int l = 0; l < L; ++l) {
    const float* Wl_l = Wl + (size_t)l * 3 * WMAT;
    const float* Wr_l = Wr + (size_t)l * 3 * WMAT;
    const float* bl_l = bl + (size_t)l * 3 * D;
    bf16* m0 = pw + (size_t)(5 * l + 0) * PM;   // Wl[l,1]
    bf16* m1 = pw + (size_t)(5 * l + 1) * PM;   // Wl[l,2]
    bf16* m2 = pw + (size_t)(5 * l + 2) * PM;   // Wr[l,1]+Wr[l,2]
    bf16* m3 = pw + (size_t)(5 * l + 3) * PM;   // Wl[l,0]
    bf16* m4 = pw + (size_t)(5 * l + 4) * PM;   // Wr[l,0]
    prepack_w_kernel<<<256, 256, 0, stream>>>(Wl_l + 1 * WMAT, nullptr, nullptr, nullptr,
                                              m0, m0 + WMAT, nullptr);
    prepack_w_kernel<<<256, 256, 0, stream>>>(Wl_l + 2 * WMAT, nullptr, nullptr, nullptr,
                                              m1, m1 + WMAT, nullptr);
    prepack_w_kernel<<<256, 256, 0, stream>>>(Wr_l + 1 * WMAT, Wr_l + 2 * WMAT,
                                              bl_l + 1 * D, bl_l + 2 * D,
                                              m2, m2 + WMAT, pbias + (size_t)(2 * l) * D);
    prepack_w_kernel<<<256, 256, 0, stream>>>(Wl_l + 0 * WMAT, nullptr, nullptr, nullptr,
                                              m3, m3 + WMAT, nullptr);
    prepack_w_kernel<<<256, 256, 0, stream>>>(Wr_l + 0 * WMAT, nullptr,
                                              bl_l + 0 * D, nullptr,
                                              m4, m4 + WMAT, pbias + (size_t)(2 * l + 1) * D);
  }

  const int scatter_blocks = (E * 32 + 255) / 256;

  for (int l = 0; l < L; ++l) {
    const float* hu_in = (l == 0) ? x_user : h1_u;
    const float* hi_in = (l == 0) ? x_item : h1_i;
    float* hu_out = (l == L - 1) ? out_u_final : h1_u;
    float* hi_out = (l == L - 1) ? out_i_final : h1_i;
    bf16* m0 = pw + (size_t)(5 * l + 0) * PM;
    bf16* m1 = pw + (size_t)(5 * l + 1) * PM;
    bf16* m2 = pw + (size_t)(5 * l + 2) * PM;
    bf16* m3 = pw + (size_t)(5 * l + 3) * PM;
    bf16* m4 = pw + (size_t)(5 * l + 4) * PM;

    // ---- user update: relu(LN( mean_iu@Wl1^T + mean_uu@Wl2^T + h_u@(Wr1+Wr2)^T + b1+b2 )) ----
    hipMemsetAsync(s_a,   0, NDmax * sizeof(float), stream);
    hipMemsetAsync(s_b,   0, NDmax * sizeof(float), stream);
    hipMemsetAsync(cnt_a, 0, (size_t)Nmax * sizeof(float), stream);
    hipMemsetAsync(cnt_b, 0, (size_t)Nmax * sizeof(float), stream);
    scatter_add_kernel<<<scatter_blocks, 256, 0, stream>>>(hi_in, ei_iu, E, s_a, cnt_a);
    scatter_add_kernel<<<scatter_blocks, 256, 0, stream>>>(hu_in, ei_uu, E, s_b, cnt_b);
    sage_gemm_ln_kernel<<<(NU + 15) / 16, 256, 0, stream>>>(
        s_a, cnt_a, m0, m0 + WMAT,
        s_b, cnt_b, m1, m1 + WMAT,
        hu_in,      m2, m2 + WMAT,
        pbias + (size_t)(2 * l) * D,
        ln_g + (size_t)(l * 2 + 0) * D, ln_b + (size_t)(l * 2 + 0) * D,
        hu_out, NU);

    // ---- item update: relu(LN( mean_ui@Wl0^T + h_i@Wr0^T + b0 )) ----
    hipMemsetAsync(s_a,   0, NDmax * sizeof(float), stream);
    hipMemsetAsync(cnt_a, 0, (size_t)Nmax * sizeof(float), stream);
    scatter_add_kernel<<<scatter_blocks, 256, 0, stream>>>(hu_in, ei_ui, E, s_a, cnt_a);
    sage_gemm_ln_kernel<<<(NI + 15) / 16, 256, 0, stream>>>(
        s_a, cnt_a, m3, m3 + WMAT,
        nullptr, nullptr, nullptr, nullptr,
        hi_in,      m4, m4 + WMAT,
        pbias + (size_t)(2 * l + 1) * D,
        ln_g + (size_t)(l * 2 + 1) * D, ln_b + (size_t)(l * 2 + 1) * D,
        hi_out, NI);
  }
}
